// LSTMBody_50835232915719
// MI455X (gfx1250) — matrix operand order repaired and checked by reading, not compiled, not run on hardware
//
#include <hip/hip_runtime.h>

// LSTM (B=256, T=512, I=512, H=1024) for gfx1250 / MI455X.
// One fused GEMM per timestep: gates[256,4096] = [x_t | h][256,1536] @ Wcat[1536,4096]
// via v_wmma_f32_16x16x32_bf16 (fp32 accumulate). Weight columns pre-permuted so each
// 128-col block tile holds gates {i,f,g,o} for the same 32 h-indices -> cell update is
// fused into the GEMM kernel. Weights (12 MiB bf16) stay L2-resident; h double-buffered
// bf16; c (fp32) lives in d_out. Staging uses global_load_async_to_lds_b128 (ASYNCcnt)
// for bf16 copies; only the x half needs a VALU f32->bf16 convert.

#define B_SZ   256
#define T_SZ   512
#define I_SZ   512
#define H_SZ   1024
#define FOURH  4096
#define KTOT   1536   // I + H

typedef __attribute__((ext_vector_type(16))) __bf16 v16bf;
typedef __attribute__((ext_vector_type(8)))  float  v8f;

__device__ __forceinline__ unsigned short f2bf(float f) {
  union { float f; unsigned u; } x; x.f = f;
  unsigned r = x.u + 0x7FFFu + ((x.u >> 16) & 1u);   // round-to-nearest-even
  return (unsigned short)(r >> 16);
}

__device__ __forceinline__ unsigned lds_off(const void* p) {
  return (unsigned)(uintptr_t)p;   // low 32 bits of flat shared address = LDS byte offset
}

// Build Wcat^T in bf16: Wbt[n'][k], n' permuted so block-local cols 0..127 =
// {gate i,f,g,o} x 32 h-indices. Wcat[k][n] = (k<I ? W_ih[n][k] : W_hh[n][k-I]).
__global__ void prep_weights(const float* __restrict__ Wih,
                             const float* __restrict__ Whh,
                             unsigned short* __restrict__ Wbt) {
  int idx = blockIdx.x * blockDim.x + threadIdx.x;   // 4096*1536
  int np = idx / KTOT;
  int k  = idx - np * KTOT;
  int gate = (np & 127) >> 5;
  int hblk = np >> 7;
  int hi   = np & 31;
  int orig = gate * H_SZ + hblk * 32 + hi;
  float v = (k < I_SZ) ? Wih[(long)orig * I_SZ + k]
                       : Whh[(long)orig * H_SZ + (k - I_SZ)];
  Wbt[idx] = f2bf(v);
}

// c = c0 (fp32, lives in d_out); h buffers = 0 (bf16); permuted combined bias.
__global__ void prep_state(const float* __restrict__ bih, const float* __restrict__ bhh,
                           const float* __restrict__ c0, float* __restrict__ c,
                           float* __restrict__ bias,
                           unsigned short* __restrict__ hA, unsigned short* __restrict__ hB) {
  int idx = blockIdx.x * blockDim.x + threadIdx.x;   // 256*1024
  c[idx]  = c0[idx];
  hA[idx] = 0;
  hB[idx] = 0;
  if (idx < FOURH) {
    int gate = (idx & 127) >> 5, hblk = idx >> 7, hi = idx & 31;
    int orig = gate * H_SZ + hblk * 32 + hi;
    bias[idx] = bih[orig] + bhh[orig];
  }
}

__global__ __launch_bounds__(256, 2)
void lstm_step(const float* __restrict__ x,
               const unsigned short* __restrict__ Wbt,
               const float* __restrict__ bias,
               const unsigned short* __restrict__ h_in,
               unsigned short* __restrict__ h_out,
               float* __restrict__ c, int t) {
  __shared__ __align__(16) unsigned char smem[32768];
  unsigned short* Alds = (unsigned short*)smem;            // [64][32] bf16 (4 KB)
  unsigned short* Blds = (unsigned short*)(smem + 4096);   // [128][32] bf16, n-major (8 KB)
  float*          Glds = (float*)smem;                     // [64][128] f32 gates (32 KB, reused)

  const int tid  = threadIdx.x;
  const int lane = tid & 31;
  const int wave = tid >> 5;
  // 2(M) x 4(N) waves; each wave owns a 32x32 tile -> 2 A frags x 2 B frags, 4 WMMA/chunk
  const int m0 = (wave >> 2) * 32;
  const int n0 = (wave & 3) * 32;
  const int b0 = blockIdx.y * 64;       // batch rows
  const int nBlk = blockIdx.x;          // 0..31 -> h-indices [nBlk*32, nBlk*32+32)

  const unsigned short* Wsrc = Wbt + (long)(nBlk * 128) * KTOT;

  // staging-role indices
  const int nrow = tid >> 1;            // B tile: 128 rows x 64B
  const int koffB = (tid & 1) * 16;     // elements
  const int arow = tid >> 2;            // A tile: 64 rows x 64B
  const int qA   = tid & 3;             // 8 elements each

  const unsigned ldsB = lds_off(Blds + nrow * 32 + koffB);
  const unsigned ldsA = lds_off(Alds + arow * 32 + qA * 8);

  v8f acc[2][2] = {};

  for (int kc = 0; kc < KTOT / 32; ++kc) {
    const int k0 = kc * 32;

    // ---- stage B tile (weights) via async global->LDS (32B per thread) ----
    {
      const unsigned short* gB = Wsrc + (long)nrow * KTOT + k0 + koffB;
      asm volatile("global_load_async_to_lds_b128 %0, %1, off\n\t"
                   "global_load_async_to_lds_b128 %0, %1, off offset:16"
                   :: "v"(ldsB), "v"(gB) : "memory");
    }
    // ---- stage A tile: x half converts f32->bf16 (VALU), h half async copy ----
    if (k0 < I_SZ) {
      const float* src = x + ((long)(b0 + arow) * T_SZ + t) * I_SZ + k0 + qA * 8;
      float4 f0 = *(const float4*)src;
      float4 f1 = *(const float4*)(src + 4);
      unsigned short* dst = Alds + arow * 32 + qA * 8;
      dst[0]=f2bf(f0.x); dst[1]=f2bf(f0.y); dst[2]=f2bf(f0.z); dst[3]=f2bf(f0.w);
      dst[4]=f2bf(f1.x); dst[5]=f2bf(f1.y); dst[6]=f2bf(f1.z); dst[7]=f2bf(f1.w);
    } else {
      const unsigned short* gA = h_in + (long)(b0 + arow) * H_SZ + (k0 - I_SZ) + qA * 8;
      asm volatile("global_load_async_to_lds_b128 %0, %1, off"
                   :: "v"(ldsA), "v"(gA) : "memory");
    }
    // prefetch next weight chunk
    if (kc + 1 < KTOT / 32) {
      const unsigned short* pf = Wsrc + (long)nrow * KTOT + k0 + 32 + koffB;
      __builtin_prefetch((const void*)pf, 0, 1);
    }
    asm volatile("s_wait_asynccnt 0x0" ::: "memory");
    __syncthreads();

    // ---- A fragments (16x32 bf16, contiguous k = 16*(lane/16) + s, same order as B) ----
    union { unsigned u[8]; v16bf v; } af[2];
    {
      int kb = (lane >= 16) ? 16 : 0;
      #pragma unroll
      for (int am = 0; am < 2; ++am) {
        int m = m0 + am * 16 + (lane & 15);
        #pragma unroll
        for (int j = 0; j < 4; ++j)
          af[am].u[j]     = *(const unsigned*)(Alds + m * 32 + kb + 2 * j);
        #pragma unroll
        for (int j = 0; j < 4; ++j)
          af[am].u[4 + j] = *(const unsigned*)(Alds + m * 32 + 8 + kb + 2 * j);
      }
    }
    // ---- 2 B fragments, each feeding 2 WMMAs ----
    const int kb2 = (lane >= 16) ? 16 : 0;
    #pragma unroll
    for (int bn = 0; bn < 2; ++bn) {
      union { unsigned u[8]; v16bf v; } bfr;
      int n = n0 + bn * 16 + (lane & 15);
      #pragma unroll
      for (int r = 0; r < 8; ++r)
        bfr.u[r] = *(const unsigned*)(Blds + n * 32 + kb2 + 2 * r);
      #pragma unroll
      for (int am = 0; am < 2; ++am)
        acc[am][bn] = __builtin_amdgcn_wmma_f32_16x16x32_bf16(
            false, af[am].v, false, bfr.v, (short)0, acc[am][bn], false, false);
    }
    __syncthreads();
  }

  // ---- spill accumulators (+bias) to LDS gate tile [64][128] ----
  #pragma unroll
  for (int am = 0; am < 2; ++am) {
    #pragma unroll
    for (int bn = 0; bn < 2; ++bn) {
      int nl = n0 + bn * 16 + (lane & 15);
      float bv = bias[nBlk * 128 + nl];
      int mb = (lane >= 16) ? 8 : 0;      // C layout: VGPR v -> M = v (+8 for upper lanes)
      #pragma unroll
      for (int v = 0; v < 8; ++v)
        Glds[(m0 + am * 16 + mb + v) * 128 + nl] = acc[am][bn][v] + bv;
    }
  }
  __syncthreads();

  // ---- fused LSTM cell update: 64 rows x 32 h-cols, 8 elems/thread ----
  #pragma unroll
  for (int it = 0; it < 8; ++it) {
    int p   = tid + it * 256;
    int row = p >> 5, hi = p & 31;
    int b   = b0 + row;
    int hg  = nBlk * 32 + hi;
    float gi = Glds[row * 128 + hi];
    float gf = Glds[row * 128 + 32 + hi];
    float gg = Glds[row * 128 + 64 + hi];
    float go = Glds[row * 128 + 96 + hi];
    long  ci = (long)b * H_SZ + hg;
    float cold = c[ci];
    float si = 1.f / (1.f + __expf(-gi));
    float sf = 1.f / (1.f + __expf(-gf));
    float so = 1.f / (1.f + __expf(-go));
    float cn = sf * cold + si * tanhf(gg);
    float hn = so * tanhf(cn);
    c[ci] = cn;                 // fp32 cell state lives in d_out
    h_out[ci] = f2bf(hn);       // bf16 hidden for next step's A matrix
  }
}

extern "C" void kernel_launch(void* const* d_in, const int* in_sizes, int n_in,
                              void* d_out, int out_size, void* d_ws, size_t ws_size,
                              hipStream_t stream) {
  const float* x   = (const float*)d_in[0];
  const float* Wih = (const float*)d_in[1];
  const float* Whh = (const float*)d_in[2];
  const float* bih = (const float*)d_in[3];
  const float* bhh = (const float*)d_in[4];
  const float* c0  = (const float*)d_in[5];
  float* c = (float*)d_out;   // final cell state, [256][1024] f32

  // workspace: Wbt (12 MiB) | bias (16 KB) | hA (512 KB) | hB (512 KB)
  unsigned char* ws = (unsigned char*)d_ws;
  unsigned short* Wbt = (unsigned short*)ws;
  size_t off = (size_t)FOURH * KTOT * sizeof(unsigned short);
  float* bias = (float*)(ws + off); off += FOURH * sizeof(float);
  off = (off + 255) & ~(size_t)255;
  unsigned short* hA = (unsigned short*)(ws + off); off += (size_t)B_SZ * H_SZ * 2;
  off = (off + 255) & ~(size_t)255;
  unsigned short* hB = (unsigned short*)(ws + off);

  prep_weights<<<(FOURH * KTOT) / 256, 256, 0, stream>>>(Wih, Whh, Wbt);
  prep_state<<<(B_SZ * H_SZ) / 256, 256, 0, stream>>>(bih, bhh, c0, c, bias, hA, hB);

  dim3 grid(FOURH / 128, B_SZ / 64);   // 32 x 4 blocks, 256 threads (8 wave32) each
  for (int t = 0; t < T_SZ; ++t) {
    const unsigned short* hin  = (t & 1) ? hB : hA;
    unsigned short*       hout = (t & 1) ? hA : hB;
    lstm_step<<<grid, 256, 0, stream>>>(x, Wbt, bias, hin, hout, c, t);
  }
}
